// HybridQLSTM_65481071403560
// MI455X (gfx1250) — matrix operand order repaired and checked by reading, not compiled, not run on hardware
//
#include <hip/hip_runtime.h>

// ---------------------------------------------------------------------------
// QLSTM on MI455X (gfx1250): persistent scan with REGISTER-RESIDENT weights.
//   S=4096, B=1, H=2048, gates=4H=8192.
//   - bf16 weight matrix (33.5 MB) is partitioned so each wave's slice is
//     exactly 128 VGPRs (16 B-fragments); loaded ONCE before the step loop.
//   - Weight columns are permuted so block nt owns all 4 gates of
//     h in [4*nt, 4*nt+4): pointwise LSTM update is local & non-redundant.
//   - Only hx (4 KB bf16, double-buffered) crosses blocks, via ONE two-level
//     atomic tree barrier per step.
// ---------------------------------------------------------------------------

typedef __attribute__((ext_vector_type(16))) __bf16 v16bf;
typedef __attribute__((ext_vector_type(8)))  float  v8f;

#define S_LEN 4096
#define HDIM  2048
#define GDIM  8192          // 4*H
#define NTILE 512           // GDIM/16 column tiles (one per block)
#define KTILE 64            // HDIM/32 k tiles
#define NBLK  512
#define TPB   128           // 4 waves (wave32)
#define NGRP  32            // barrier tree: 32 groups of 16 blocks

// barrier array indices (u32 units; one counter per 128B line)
#define BAR_GEN   0
#define BAR_ROOT  32
#define BAR_GRP(g) (64 + (g) * 32)
#define BAR_WORDS (64 + NGRP * 32)

// workspace layout (bytes)
static constexpr size_t WPACK_OFF = 0;                                   // bf16 packed W
static constexpr size_t WPACK_SZ  = (size_t)NTILE * KTILE * 32 * 16 * 2; // 33.5 MB
static constexpr size_t HXG_OFF   = WPACK_OFF + WPACK_SZ;                // bf16[2][2048]
static constexpr size_t CS_OFF    = HXG_OFF + 2 * HDIM * 2;              // f32[4096]
static constexpr size_t BAR_OFF   = CS_OFF + S_LEN * 4;                  // u32[BAR_WORDS]

__device__ __forceinline__ unsigned short f2bf(float x) {
    unsigned u = __float_as_uint(x);
    unsigned r = (u >> 16) & 1u;
    u += 0x7FFFu + r;                 // round-to-nearest-even
    return (unsigned short)(u >> 16);
}

__device__ __forceinline__ float sigmoidf_(float x) {
    return 1.0f / (1.0f + __expf(-x));
}

// column permutation: packed column nt*16 + lp  (lp = 0..15) maps to original
// gate column  q*HDIM + (nt*4 + hh)  with hh = lp>>2, q = lp&3  (f,i,g,o).
__device__ __forceinline__ int orig_col(int nt, int lp) {
    return (lp & 3) * HDIM + nt * 4 + (lp >> 2);
}

// ---------------------------------------------------------------------------
// Phase 0a: conv preprocess.  conv_scalar[s] = sigmoid(inputs[s] . conv_w + b)
// ---------------------------------------------------------------------------
__global__ void qlstm_conv_pre(const float* __restrict__ in,
                               const float* __restrict__ cw,
                               const float* __restrict__ cb,
                               float* __restrict__ cs) {
    int s = blockIdx.x * blockDim.x + threadIdx.x;
    if (s < S_LEN) {
        float x = cb[0];
        #pragma unroll
        for (int j = 0; j < 4; ++j) x += in[s * 4 + j] * cw[j];
        cs[s] = sigmoidf_(x);
    }
}

// ---------------------------------------------------------------------------
// Phase 0b: pack w_gates rows 1..2048 (bf16) into WMMA B-fragment tile order
// with the gate-interleaved column permutation.
// B (32x16 bf16): lane l holds column n = l&15; K-half = (l>>4)*16.
// ---------------------------------------------------------------------------
__global__ void qlstm_pack_w(const float* __restrict__ wg,
                             unsigned short* __restrict__ wp) {
    size_t t = (size_t)blockIdx.x * blockDim.x + threadIdx.x; // [0, 512*64*32)
    int l  = (int)(t & 31);
    int kt = (int)((t >> 5) & 63);
    int nt = (int)(t >> 11);
    int n     = orig_col(nt, l & 15);
    int kbase = kt * 32 + ((l >> 4) << 4);
    unsigned short tmp[16];
    #pragma unroll
    for (int j = 0; j < 16; ++j) {
        float v = wg[(size_t)(kbase + j + 1) * GDIM + n]; // +1 skips c_t row
        tmp[j] = f2bf(v);
    }
    unsigned short* dst = wp + t * 16;
    *(uint4*)(dst)     = *(const uint4*)(tmp);
    *(uint4*)(dst + 8) = *(const uint4*)(tmp + 8);
}

// ---------------------------------------------------------------------------
// Phase 0c: init barrier tree.
// ---------------------------------------------------------------------------
__global__ void qlstm_init(unsigned int* __restrict__ bar) {
    int i = blockIdx.x * blockDim.x + threadIdx.x;
    if (i < BAR_WORDS) bar[i] = 0;
}

// ---------------------------------------------------------------------------
// Persistent LSTM scan kernel.  Block nt owns gate tile nt (= all 4 gates of
// h in [4nt, 4nt+4)).  Weights live in VGPRs for the whole scan.
// ---------------------------------------------------------------------------
__global__ void __launch_bounds__(TPB, 1)
qlstm_scan(const float* __restrict__ w_gates,   // row 0 = c_t weights (fp32)
           const float* __restrict__ b_gates,
           const unsigned short* __restrict__ wpack,
           const float* __restrict__ conv_s,
           unsigned short* __restrict__ hxg,    // bf16[2][2048] double-buffered
           unsigned int* __restrict__ bar,
           float* __restrict__ out) {
    const int tid = threadIdx.x;
    const int w   = tid >> 5;        // wave id 0..3
    const int l   = tid & 31;        // lane id
    const int nt  = blockIdx.x;      // column tile
    const int grp = nt >> 4;         // barrier group (16 blocks each)

    __shared__ __align__(16) unsigned short hx_lds[HDIM]; // hx (bf16)
    __shared__ float red[4][16];
    __shared__ float gsh[16];

    for (int i = tid; i < HDIM; i += TPB) hx_lds[i] = 0;

    // per-step invariants for the gates epilogue (wave 0, lanes 0..15)
    float w0n = 0.0f, bbn = 0.0f;
    if (w == 0 && l < 16) {
        int n = orig_col(nt, l);
        w0n = w_gates[n];            // c_t weight (row 0)
        bbn = b_gates[n];
    }
    float cx_l = 0.0f;               // cell state, wave0 lanes 0..3 only

    // A fragments (16x32 bf16, row 0 only): zero ONCE; exec-masked LDS loads
    // into the same registers leave lanes != 0/16 at zero forever.
    union AF { v16bf v; uint4 q[2]; };
    union BF { v16bf v; uint4 q[2]; };
    AF a0, a1;
    a0.q[0] = make_uint4(0, 0, 0, 0); a0.q[1] = make_uint4(0, 0, 0, 0);
    a1.q[0] = make_uint4(0, 0, 0, 0); a1.q[1] = make_uint4(0, 0, 0, 0);
    const bool aload = (l == 0) || (l == 16);
    const unsigned short* hx_base = hx_lds + ((l >> 4) << 4);
    const int kt0 = w * 16;

#define LOAD_A(dst, kt)                                                     \
    do { if (aload) {                                                       \
        const unsigned short* hp_ = hx_base + (kt) * 32;                    \
        (dst).q[0] = *(const uint4*)(hp_);                                  \
        (dst).q[1] = *(const uint4*)(hp_ + 8);                              \
    } } while (0)

    // ---- preload this wave's ENTIRE weight slice into VGPRs (128 regs) ----
    const unsigned short* wp_nt = wpack + (size_t)nt * KTILE * 32 * 16;
    BF Bf[16];
    #pragma unroll
    for (int i = 0; i < 16; ++i) {
        const unsigned short* bp = wp_nt + ((size_t)(kt0 + i) * 32 + l) * 16;
        Bf[i].q[0] = *(const uint4*)(bp);
        Bf[i].q[1] = *(const uint4*)(bp + 8);
    }

    __syncthreads();

    for (int s = 0; s < S_LEN; ++s) {
        unsigned short* hxp = hxg + (size_t)(s & 1) * HDIM;

        // ---- phase A: 16 WMMAs against register-resident B fragments ----
        union { v8f v; float f[8]; } acc0, acc1;
        #pragma unroll
        for (int i = 0; i < 8; ++i) { acc0.f[i] = 0.0f; acc1.f[i] = 0.0f; }

        LOAD_A(a0, kt0);
        LOAD_A(a1, kt0 + 1);
        #pragma unroll
        for (int i = 0; i < 16; i += 2) {
            acc0.v = __builtin_amdgcn_wmma_f32_16x16x32_bf16(
                        false, a0.v, false, Bf[i].v, (short)0, acc0.v, false, false);
            if (i + 2 < 16) LOAD_A(a0, kt0 + i + 2);
            acc1.v = __builtin_amdgcn_wmma_f32_16x16x32_bf16(
                        false, a1.v, false, Bf[i + 1].v, (short)0, acc1.v, false, false);
            if (i + 3 < 16) LOAD_A(a1, kt0 + i + 3);
        }

        // C layout: M=0 row = element 0, lanes 0..15 (N = lane).
        if (l < 16) red[w][l] = acc0.f[0] + acc1.f[0];
        __syncthreads();
        if (w == 0 && l < 16)
            gsh[l] = red[0][l] + red[1][l] + red[2][l] + red[3][l]
                   + conv_s[s] * w0n + bbn;           // c_t row + bias

        // ---- pointwise LSTM update: local, once per h (wave0 lanes 0..3) --
        if (w == 0 && l < 4) {
            int   h  = nt * 4 + l;
            float f  = sigmoidf_(gsh[4 * l + 0]);
            float i_ = sigmoidf_(gsh[4 * l + 1]);
            float g  = tanhf    (gsh[4 * l + 2]);
            float o  = sigmoidf_(gsh[4 * l + 3]);
            float c  = f * cx_l + i_ * g;
            float hn = o * tanhf(c);
            cx_l = c;
            hxp[h] = f2bf(hn);                         // publish hx (bf16)
            out[(size_t)s * HDIM + h] = hn;
            if (s == S_LEN - 1) {
                out[(size_t)S_LEN * HDIM + h]        = hn; // hx
                out[(size_t)S_LEN * HDIM + HDIM + h] = c;  // cx
            }
        }

        // ---- single grid barrier per step (two-level tree) ----
        __threadfence();
        __syncthreads();
        if (tid == 0) {
            unsigned g = __hip_atomic_load(&bar[BAR_GEN], __ATOMIC_RELAXED,
                                           __HIP_MEMORY_SCOPE_AGENT);
            if (__hip_atomic_fetch_add(&bar[BAR_GRP(grp)], 1u, __ATOMIC_ACQ_REL,
                                       __HIP_MEMORY_SCOPE_AGENT) == 15u) {
                __hip_atomic_store(&bar[BAR_GRP(grp)], 0u, __ATOMIC_RELAXED,
                                   __HIP_MEMORY_SCOPE_AGENT);
                if (__hip_atomic_fetch_add(&bar[BAR_ROOT], 1u, __ATOMIC_ACQ_REL,
                                           __HIP_MEMORY_SCOPE_AGENT) == NGRP - 1u) {
                    __hip_atomic_store(&bar[BAR_ROOT], 0u, __ATOMIC_RELAXED,
                                       __HIP_MEMORY_SCOPE_AGENT);
                    __hip_atomic_fetch_add(&bar[BAR_GEN], 1u, __ATOMIC_RELEASE,
                                           __HIP_MEMORY_SCOPE_AGENT);
                }
            }
            while (__hip_atomic_load(&bar[BAR_GEN], __ATOMIC_ACQUIRE,
                                     __HIP_MEMORY_SCOPE_AGENT) == g)
                __builtin_amdgcn_s_sleep(1);
        }
        __syncthreads();
        __threadfence();

        // ---- refill block-local hx from published vector (4 KB) ----
        // (hxg double-buffered on parity -> no WAR across the single barrier)
        {
            const uint4* src = (const uint4*)(hxp);    // 256 uint4
            uint4* dst = (uint4*)hx_lds;
            dst[tid]       = src[tid];
            dst[tid + 128] = src[tid + 128];
        }
        __syncthreads();   // hx_lds ready for next step's A fragments
    }
#undef LOAD_A
}

// ---------------------------------------------------------------------------
extern "C" void kernel_launch(void* const* d_in, const int* in_sizes, int n_in,
                              void* d_out, int out_size, void* d_ws, size_t ws_size,
                              hipStream_t stream) {
    const float* inputs  = (const float*)d_in[0];  // (4096,1,4)
    const float* conv_w  = (const float*)d_in[1];  // (4,)
    const float* conv_b  = (const float*)d_in[2];  // (1,)
    const float* w_gates = (const float*)d_in[3];  // (2049,8192)
    const float* b_gates = (const float*)d_in[4];  // (8192,)
    float* out = (float*)d_out;                    // stacked | hx | cx

    char* ws = (char*)d_ws;
    unsigned short* wpack = (unsigned short*)(ws + WPACK_OFF);
    unsigned short* hxg   = (unsigned short*)(ws + HXG_OFF);
    float*          cs    = (float*)(ws + CS_OFF);
    unsigned int*   bar   = (unsigned int*)(ws + BAR_OFF);

    // per-call preprocessing, stream-ordered
    qlstm_conv_pre<<<S_LEN / 256, 256, 0, stream>>>(inputs, conv_w, conv_b, cs);
    qlstm_pack_w<<<(NTILE * KTILE * 32) / 256, 256, 0, stream>>>(w_gates, wpack);
    qlstm_init<<<(BAR_WORDS + 255) / 256, 256, 0, stream>>>(bar);

    // persistent sequential scan (weights register-resident, 1 barrier/step)
    qlstm_scan<<<NBLK, TPB, 0, stream>>>(w_gates, b_gates, wpack, cs,
                                         hxg, bar, out);
}
